// MSARowAttentionWithPairBias_66520453481104
// MI455X (gfx1250) — matrix operand order repaired and checked by reading, not compiled, not run on hardware
//
#include <hip/hip_runtime.h>
#include <hip/hip_bf16.h>

typedef __attribute__((ext_vector_type(16))) _Float16 v16h;
typedef __attribute__((ext_vector_type(8)))  float    v8f;

#define CM   256
#define CZ   128
#define NH   8
#define HD   32
#define LLEN 256
#define NSEQ 64
#define NEGVAL  (-10000.0f)
#define LN_EPS  1e-5f
#define SCALE_F 0.17677669529663687f   // 32^-0.5

// A-fragment (16-bit, 16x32 MxK) per-lane column index for element e.
__device__ __forceinline__ int a_col(int e, int hi) {
  return e + hi * 16;
}

// ---------------------------------------------------------------- LayerNorm
__global__ void ln_kernel(const float* __restrict__ m,
                          const float* __restrict__ g,
                          const float* __restrict__ b,
                          _Float16* __restrict__ X) {
  const int row = blockIdx.x;          // 0..16383  (n*256 + l)
  const int tid = threadIdx.x;         // 0..255 == channel
  __shared__ float red[8];
  const float x = m[(size_t)row * CM + tid];

  float s = x;
  #pragma unroll
  for (int off = 16; off > 0; off >>= 1) s += __shfl_xor(s, off, 32);
  if ((tid & 31) == 0) red[tid >> 5] = s;
  __syncthreads();
  float tot = 0.f;
  #pragma unroll
  for (int i = 0; i < 8; ++i) tot += red[i];
  const float mu = tot * (1.0f / CM);
  const float d  = x - mu;
  __syncthreads();

  s = d * d;
  #pragma unroll
  for (int off = 16; off > 0; off >>= 1) s += __shfl_xor(s, off, 32);
  if ((tid & 31) == 0) red[tid >> 5] = s;
  __syncthreads();
  tot = 0.f;
  #pragma unroll
  for (int i = 0; i < 8; ++i) tot += red[i];
  const float var = tot * (1.0f / CM);

  const float xn = d * rsqrtf(var + LN_EPS) * g[tid] + b[tid];
  X[(size_t)row * CM + tid] = (_Float16)xn;
}

// ------------------------------------------------- Q/K/V projection (WMMA)
// One wave computes one 16x16 tile of one projection; K=256 in 8 chunks.
__global__ void qkv_kernel(const _Float16* __restrict__ X,
                           const float* __restrict__ Wq, const float* __restrict__ bq,
                           const float* __restrict__ Wk, const float* __restrict__ bk,
                           const float* __restrict__ Wv, const float* __restrict__ bv,
                           _Float16* __restrict__ Q, _Float16* __restrict__ K,
                           _Float16* __restrict__ V) {
  const int lane = threadIdx.x & 31;
  const int wave = threadIdx.x >> 5;
  const int tile = blockIdx.x * 8 + wave;      // 0..49151
  const int proj = tile / 16384;               // 0=q 1=k 2=v
  const int t    = tile - proj * 16384;
  const int rt   = t >> 4;                     // row tile 0..1023
  const int ct   = t & 15;                     // col tile 0..15
  const int mloc = lane & 15;
  const int hi   = lane >> 4;

  const float* W; const float* bias; _Float16* dst;
  if (proj == 0)      { W = Wq; bias = bq; dst = Q; }
  else if (proj == 1) { W = Wk; bias = bk; dst = K; }
  else                { W = Wv; bias = bv; dst = V; }

  const int j = ct * 16 + mloc;                // output feature (B column)
  v8f acc = {};
  #pragma unroll
  for (int kc = 0; kc < 8; ++kc) {
    v16h a, bb;
    #pragma unroll
    for (int e = 0; e < 16; ++e) {
      a[e]  = X[(size_t)(rt * 16 + mloc) * CM + kc * 32 + a_col(e, hi)];
      bb[e] = (_Float16)W[(size_t)j * CM + kc * 32 + hi * 16 + e];
    }
    acc = __builtin_amdgcn_wmma_f32_16x16x32_f16(false, a, false, bb,
                                                 (short)0, acc, false, false);
  }
  const float bj = bias[j];
  const int h = j >> 5, d = j & 31;
  #pragma unroll
  for (int r = 0; r < 8; ++r) {
    const int i = rt * 16 + r + 8 * hi;        // global row (n*256 + l)
    const int n = i >> 8, l = i & 255;
    dst[((size_t)(n * NH + h) * LLEN + l) * HD + d] = (_Float16)(acc[r] + bj);
  }
}

// ------------------------------------- pair bias z@Wpb^T as WMMA (N padded)
// GEMM (65536 x 128) @ (128 x 8->16). One wave per 16-row tile, 4 k32 WMMAs.
__global__ void pair_bias_kernel(const float* __restrict__ z,
                                 const float* __restrict__ Wpb,
                                 float* __restrict__ bias) {
  const int lane = threadIdx.x & 31;
  const int wave = threadIdx.x >> 5;
  const int rt   = blockIdx.x * 8 + wave;      // 0..4095 (16 pairs each)
  const int mloc = lane & 15, hi = lane >> 4;

  v8f acc = {};
  #pragma unroll
  for (int kc = 0; kc < 4; ++kc) {
    v16h a, bb;
    #pragma unroll
    for (int e = 0; e < 16; ++e) {
      a[e]  = (_Float16)z[(size_t)(rt * 16 + mloc) * CZ + kc * 32 + a_col(e, hi)];
      bb[e] = (mloc < NH)
                ? (_Float16)Wpb[mloc * CZ + kc * 32 + hi * 16 + e]
                : (_Float16)0.f;               // zero-padded columns h>=8
    }
    acc = __builtin_amdgcn_wmma_f32_16x16x32_f16(false, a, false, bb,
                                                 (short)0, acc, false, false);
  }
  if (mloc < NH) {
    #pragma unroll
    for (int r = 0; r < 8; ++r) {
      const int p = rt * 16 + r + 8 * hi;      // (q,k) pair index
      bias[(size_t)mloc * 65536 + p] = acc[r]; // bias[h][q][k]
    }
  }
}

// ------------------------------------------------------------ attention core
__global__ void __launch_bounds__(256)
attn_kernel(const _Float16* __restrict__ Q,
            const _Float16* __restrict__ K,
            const _Float16* __restrict__ V,
            const float* __restrict__ biasT,
            const unsigned char* __restrict__ rmask,
            const unsigned char* __restrict__ mmask,
            _Float16* __restrict__ O) {
  __shared__ _Float16 Ks[LLEN * HD];           // 16 KB
  __shared__ _Float16 Vs[LLEN * HD];           // 16 KB
  __shared__ _Float16 Ps[8][16 * LLEN];        // 64 KB (per-wave P tile)

  const int nh = blockIdx.x;                   // 0..511
  const int n = nh >> 3, h = nh & 7;
  const int tid = threadIdx.x;
  const int lane = tid & 31, wave = tid >> 5;
  const int mloc = lane & 15, hi = lane >> 4;

  const size_t base = (size_t)(n * NH + h) * LLEN * HD;   // 8192 halves
  // stage K and V tiles (32-bit copies, coalesced)
  const unsigned int* Kg = (const unsigned int*)(K + base);
  const unsigned int* Vg = (const unsigned int*)(V + base);
  unsigned int* KsU = (unsigned int*)Ks;
  unsigned int* VsU = (unsigned int*)Vs;
  #pragma unroll
  for (int i = 0; i < 8; ++i) {
    KsU[tid + i * 256] = Kg[tid + i * 256];
    VsU[tid + i * 256] = Vg[tid + i * 256];
  }
  __syncthreads();

  const bool mm = mmask[n] != 0;
  const float* biasH = biasT + (size_t)h * LLEN * LLEN;

  for (int qi = 0; qi < 2; ++qi) {
    const int qt = wave * 2 + qi;              // 16-row Q tile, 0..15
    // Q A-fragment (16x32, K = head dim)
    v16h qa;
    #pragma unroll
    for (int e = 0; e < 16; ++e)
      qa[e] = Q[base + (size_t)(qt * 16 + mloc) * HD + a_col(e, hi)];

    // logits: 16 tiles of 16x16, one WMMA each (K=D=32)
    v8f s[16];
    #pragma unroll
    for (int kt = 0; kt < 16; ++kt) {
      v16h kb;
      #pragma unroll
      for (int e = 0; e < 16; ++e)
        kb[e] = Ks[(kt * 16 + mloc) * HD + hi * 16 + e];
      v8f zacc = {};
      s[kt] = __builtin_amdgcn_wmma_f32_16x16x32_f16(false, qa, false, kb,
                                                     (short)0, zacc, false, false);
    }
    // scale + pair bias + masks
    #pragma unroll
    for (int kt = 0; kt < 16; ++kt) {
      const int key = kt * 16 + mloc;
      const bool kok = mm && (rmask[key] != 0);
      #pragma unroll
      for (int r = 0; r < 8; ++r) {
        const int qrow = qt * 16 + r + 8 * hi;
        const bool ok = kok && (rmask[qrow] != 0);
        const float v = s[kt][r] * SCALE_F + biasH[(size_t)qrow * LLEN + key];
        s[kt][r] = ok ? v : NEGVAL;
      }
    }
    // softmax: rows of a tile live in one 16-lane group -> shfl width 16
    float mx[8], sm[8];
    #pragma unroll
    for (int r = 0; r < 8; ++r) {
      float m0 = -3.0e38f;
      #pragma unroll
      for (int kt = 0; kt < 16; ++kt) m0 = fmaxf(m0, s[kt][r]);
      #pragma unroll
      for (int off = 8; off > 0; off >>= 1) m0 = fmaxf(m0, __shfl_xor(m0, off, 16));
      mx[r] = m0;
    }
    #pragma unroll
    for (int r = 0; r < 8; ++r) {
      float t0 = 0.f;
      #pragma unroll
      for (int kt = 0; kt < 16; ++kt) {
        const float p = __expf(s[kt][r] - mx[r]);
        s[kt][r] = p;
        t0 += p;
      }
      #pragma unroll
      for (int off = 8; off > 0; off >>= 1) t0 += __shfl_xor(t0, off, 16);
      sm[r] = 1.0f / t0;
    }
    // normalized P -> per-wave LDS tile (row-major 16 x 256, f16)
    #pragma unroll
    for (int kt = 0; kt < 16; ++kt) {
      #pragma unroll
      for (int r = 0; r < 8; ++r) {
        const int mrow = r + 8 * hi;
        Ps[wave][mrow * LLEN + kt * 16 + mloc] = (_Float16)(s[kt][r] * sm[r]);
      }
    }
    // O = P (16x256) @ V (256x32): 2 N-tiles x 8 K-chunks
    #pragma unroll
    for (int nt = 0; nt < 2; ++nt) {
      v8f o = {};
      #pragma unroll
      for (int kc = 0; kc < 8; ++kc) {
        v16h pa, vb;
        #pragma unroll
        for (int e = 0; e < 16; ++e) {
          pa[e] = Ps[wave][mloc * LLEN + kc * 32 + a_col(e, hi)];
          vb[e] = Vs[(kc * 32 + hi * 16 + e) * HD + nt * 16 + mloc];
        }
        o = __builtin_amdgcn_wmma_f32_16x16x32_f16(false, pa, false, vb,
                                                   (short)0, o, false, false);
      }
      #pragma unroll
      for (int r = 0; r < 8; ++r) {
        const int qrow = qt * 16 + r + 8 * hi;
        const int d = nt * 16 + mloc;
        // layout: row (n*256+l), col (h*32+d)  == concat-heads for out-proj
        O[((size_t)(n * LLEN + qrow) * NH + h) * HD + d] = (_Float16)o[r];
      }
    }
  }
}

// ------------------------------------------------------- output projection
__global__ void outproj_kernel(const _Float16* __restrict__ O,
                               const float* __restrict__ Wo,
                               const float* __restrict__ bo,
                               const unsigned char* __restrict__ mmask,
                               float* __restrict__ out) {
  const int lane = threadIdx.x & 31;
  const int wave = threadIdx.x >> 5;
  const int tile = blockIdx.x * 8 + wave;      // 0..16383
  const int rt = tile >> 4, ct = tile & 15;
  const int mloc = lane & 15, hi = lane >> 4;
  const int j = ct * 16 + mloc;

  v8f acc = {};
  #pragma unroll
  for (int kc = 0; kc < 8; ++kc) {
    v16h a, bb;
    #pragma unroll
    for (int e = 0; e < 16; ++e) {
      a[e]  = O[(size_t)(rt * 16 + mloc) * CM + kc * 32 + a_col(e, hi)];
      bb[e] = (_Float16)Wo[(size_t)j * CM + kc * 32 + hi * 16 + e];
    }
    acc = __builtin_amdgcn_wmma_f32_16x16x32_f16(false, a, false, bb,
                                                 (short)0, acc, false, false);
  }
  const float bj = bo[j];
  #pragma unroll
  for (int r = 0; r < 8; ++r) {
    const int i = rt * 16 + r + 8 * hi;
    const int n = i >> 8;
    const float mk = mmask[n] ? 1.f : 0.f;
    out[(size_t)i * CM + j] = (acc[r] + bj) * mk;
  }
}

// ---------------------------------------------------------------- launcher
extern "C" void kernel_launch(void* const* d_in, const int* in_sizes, int n_in,
                              void* d_out, int out_size, void* d_ws, size_t ws_size,
                              hipStream_t stream) {
  const float* m    = (const float*)d_in[0];
  const float* z    = (const float*)d_in[1];
  const unsigned char* rmask = (const unsigned char*)d_in[2];
  const unsigned char* mmask = (const unsigned char*)d_in[3];
  const float* ln_g = (const float*)d_in[4];
  const float* ln_b = (const float*)d_in[5];
  const float* Wq   = (const float*)d_in[6];
  const float* bq   = (const float*)d_in[7];
  const float* Wk   = (const float*)d_in[8];
  const float* bk   = (const float*)d_in[9];
  const float* Wv   = (const float*)d_in[10];
  const float* bv   = (const float*)d_in[11];
  const float* Wo   = (const float*)d_in[12];
  const float* bo   = (const float*)d_in[13];
  const float* Wpb  = (const float*)d_in[14];

  char* ws = (char*)d_ws;
  _Float16* X  = (_Float16*)(ws);                       // 8 MB  LN output f16
  _Float16* Q  = (_Float16*)(ws + (size_t)(8u  << 20)); // 8 MB  [n][h][l][d]
  _Float16* K  = (_Float16*)(ws + (size_t)(16u << 20)); // 8 MB
  _Float16* V  = (_Float16*)(ws + (size_t)(24u << 20)); // 8 MB
  float*    bias = (float*)(ws + (size_t)(32u << 20));  // 2 MB  [h][q][k]
  _Float16* O  = (_Float16*)(ws + (size_t)(34u << 20)); // 8 MB  [n][l][h*d]

  ln_kernel       <<<16384, 256, 0, stream>>>(m, ln_g, ln_b, X);
  qkv_kernel      <<<6144,  256, 0, stream>>>(X, Wq, bq, Wk, bk, Wv, bv, Q, K, V);
  pair_bias_kernel<<<512,   256, 0, stream>>>(z, Wpb, bias);
  attn_kernel     <<<512,   256, 0, stream>>>(Q, K, V, bias, rmask, mmask, O);
  outproj_kernel  <<<2048,  256, 0, stream>>>(O, Wo, bo, mmask, (float*)d_out);
}
